// BayesianLSTM_59742995088078
// MI455X (gfx1250) — matrix operand order repaired and checked by reading, not compiled, not run on hardware
//
#include <hip/hip_runtime.h>
#include <hip/hip_bf16.h>

// ---------------------------------------------------------------------------
// Bayesian LSTM on gfx1250 (MI455X), bf16 WMMA with f32 accumulate.
//   bs=64, seq=512, IN=256, HS=512, 4HS=2048
// 16 WGs x 256 thr; WG owns hidden tiles {2*blk, 2*blk+1}; waves = (mt,hl).
// W tiles (192 KB) live in LDS for the whole 512-step scan (320 KB WGP LDS).
// A tiles are bf16 end-to-end: x pre-converted once, recurrent h carried as a
// bf16 global buffer, so the per-step fill is pure 16B copies — issued as
// global_load_async_to_lds_b128 (ASYNCcnt) when the builtin is available.
// Cell state c stays in VGPRs for the whole scan; one software grid barrier
// per step.
//
// Workspace layout (bytes):
//   [0,        256)       : grid barrier {counter, generation}
//   [256,      131328)    : h bf16 double buffer, 2 x [64][512]
//   [131328,   139520)    : sampled bias f32 [2048]
//   [139520,   16916736)  : x in bf16 [64][512][256]
//   [16916736, 17965312)  : W_ih packed bf16 (B-matrix tile layout)
//   [17965312, 20062464)  : W_hh packed bf16 (B-matrix tile layout)
// ---------------------------------------------------------------------------

typedef __attribute__((ext_vector_type(16))) __bf16 v16bf;
typedef __attribute__((ext_vector_type(8)))  __bf16 v8bf;
typedef __attribute__((ext_vector_type(4)))  __bf16 v4bf;
typedef __attribute__((ext_vector_type(8)))  float  v8f;
typedef __attribute__((__vector_size__(4 * sizeof(int)))) int v4i;

#define BSZ 64
#define SEQ 512
#define INF 256
#define HS  512
#define G4  2048
#define NWG 16

#define WS_BAR  0
#define WS_HBF  256
#define WS_BIAS 131328
#define WS_XBF  139520
#define WS_PIH  16916736
#define WS_PHH  17965312

#if __has_builtin(__builtin_amdgcn_global_load_async_to_lds_b128)
#define HAVE_ASYNC_LDS 1
#else
#define HAVE_ASYNC_LDS 0
#endif

// ---------------------------------------------------------------------------
__global__ void sample_bias_kernel(const float* __restrict__ mu,
                                   const float* __restrict__ rho,
                                   const float* __restrict__ eps,
                                   float* __restrict__ out, int n) {
  int i = blockIdx.x * blockDim.x + threadIdx.x;
  if (i < n) out[i] = mu[i] + __expf(rho[i]) * eps[i];
}

__global__ void cvt_x_kernel(const float* __restrict__ in,
                             __bf16* __restrict__ outp, int n4) {
  int i = blockIdx.x * blockDim.x + threadIdx.x;
  if (i < n4) {
    float4 v = ((const float4*)in)[i];
    v4bf o;
    o[0] = (__bf16)v.x; o[1] = (__bf16)v.y; o[2] = (__bf16)v.z; o[3] = (__bf16)v.w;
    ((v4bf*)outp)[i] = o;
  }
}

// Sample W = mu + exp(rho)*eps and pack into WMMA B tiles (contiguous k, same order as the A fill).
// B tile (32x16 bf16) per (kk, nt):  lane<16 -> col=nt*16+lane, K=kk*32+e
//                                    lane>=16 -> col=nt*16+lane-16, K=kk*32+16+e
__global__ void pack_weights_kernel(const float* __restrict__ mu,
                                    const float* __restrict__ rho,
                                    const float* __restrict__ eps,
                                    __bf16* __restrict__ packed, int K, int N) {
  int idx = blockIdx.x * blockDim.x + threadIdx.x;
  if (idx >= K * N) return;
  int e    = idx & 15;
  int lane = (idx >> 4) & 31;
  int tile = idx >> 9;
  int NT   = N >> 4;
  int nt   = tile % NT;
  int kk   = tile / NT;
  int k = kk * 32 + ((lane >= 16) ? 16 : 0) + e;
  int n = nt * 16 + (lane & 15);
  size_t src = (size_t)k * N + n;
  float w = mu[src] + __expf(rho[src]) * eps[src];
  packed[idx] = (__bf16)w;
}

// ---------------------------------------------------------------------------
__device__ __forceinline__ void grid_barrier(unsigned* counter, unsigned* gen) {
  __syncthreads();
  if (threadIdx.x == 0) {
    __threadfence();  // release: flush WGP-level cache
    unsigned g = __hip_atomic_load(gen, __ATOMIC_RELAXED, __HIP_MEMORY_SCOPE_AGENT);
    unsigned prev = __hip_atomic_fetch_add(counter, 1u, __ATOMIC_ACQ_REL,
                                           __HIP_MEMORY_SCOPE_AGENT);
    if (prev == NWG - 1u) {
      __hip_atomic_store(counter, 0u, __ATOMIC_RELAXED, __HIP_MEMORY_SCOPE_AGENT);
      __hip_atomic_store(gen, g + 1u, __ATOMIC_RELEASE, __HIP_MEMORY_SCOPE_AGENT);
    } else {
      while (__hip_atomic_load(gen, __ATOMIC_ACQUIRE, __HIP_MEMORY_SCOPE_AGENT) == g)
        __builtin_amdgcn_s_sleep(1);
    }
    __threadfence();  // acquire: invalidate WGP-level cache
  }
  __syncthreads();
}

// Copy 16B global -> LDS.  Async (ASYNCcnt, no VGPR round-trip) if available.
// ROCm clang-22 signature takes generic v4i* pointers (probe: round-4 error).
__device__ __forceinline__ void copy16_g2l(const __bf16* __restrict__ g,
                                           __bf16* __restrict__ l) {
#if HAVE_ASYNC_LDS
  __builtin_amdgcn_global_load_async_to_lds_b128((v4i*)g, (v4i*)l, 0, 0);
#else
  *(v8bf*)l = *(const v8bf*)g;
#endif
}

__device__ __forceinline__ void wait_fill() {
#if HAVE_ASYNC_LDS
#if __has_builtin(__builtin_amdgcn_s_wait_asynccnt)
  __builtin_amdgcn_s_wait_asynccnt(0);
#else
  asm volatile("s_wait_asynccnt 0x0" ::: "memory");
#endif
#endif
}

// ---------------------------------------------------------------------------
__global__ void __launch_bounds__(256)
lstm_scan_kernel(const __bf16* __restrict__ xbf,
                 const __bf16* __restrict__ pb_ih,
                 const __bf16* __restrict__ pb_hh,
                 const float* __restrict__ bias,
                 __bf16* __restrict__ hbf,
                 unsigned* __restrict__ bar,
                 float* __restrict__ out) {
  // Scan-resident weight tiles + per-step A staging: 64+128+96 = 288 KB LDS.
  __shared__ __bf16 sBih[64 * 512];      // [ (hl*4+g)*8  + kk ][lane*16]
  __shared__ __bf16 sBhh[128 * 512];     // [ (hl*4+g)*16 + kk ][lane*16]
  __shared__ __bf16 sA[4 * 24 * 512];    // [ mt*24 + t ][lane*16]

  const unsigned tid   = threadIdx.x;
  const unsigned lane  = tid & 31u;
  const unsigned wave  = tid >> 5;
  const unsigned mt    = wave & 3u;               // batch tile 0..3
  const unsigned hl    = wave >> 2;               // 0..1
  const unsigned ht0   = blockIdx.x * 2u;
  const unsigned ht    = ht0 + hl;                // hidden tile 0..31
  const unsigned rlo   = lane & 15u;
  const unsigned khalf = (lane >= 16u) ? 16u : 0u;
  const unsigned brow  = mt * 16u + rlo;          // batch row (fill)
  const unsigned col   = ht * 16u + rlo;          // hidden col (C/D)
  const unsigned crow0 = mt * 16u + ((lane >= 16u) ? 8u : 0u);

  // ---- one-time: pull this WG's W tiles into LDS (resident for the scan) --
  for (int u = (int)tid; u < 64 * 64; u += 256) {         // W_ih: 64 KB
    int t = u >> 6, ch = u & 63;
    int fhl = t >> 5, g = (t >> 3) & 3, kk = t & 7;
    int nt = g * 32 + (int)ht0 + fhl;
    *(float4*)(&sBih[(size_t)t * 512 + (size_t)ch * 8]) =
        *(const float4*)(pb_ih + ((size_t)(kk * 128 + nt) * 512 + (size_t)ch * 8));
  }
  for (int u = (int)tid; u < 128 * 64; u += 256) {        // W_hh: 128 KB
    int t = u >> 6, ch = u & 63;
    int fhl = t >> 6, g = (t >> 4) & 3, kk = t & 15;
    int nt = g * 32 + (int)ht0 + fhl;
    *(float4*)(&sBhh[(size_t)t * 512 + (size_t)ch * 8]) =
        *(const float4*)(pb_hh + ((size_t)(kk * 128 + nt) * 512 + (size_t)ch * 8));
  }

  float bv[4];
#pragma unroll
  for (int g = 0; g < 4; ++g) bv[g] = bias[g * HS + ht * 16 + rlo];

  v8f c, hlast;
#pragma unroll
  for (int j = 0; j < 8; ++j) { c[j] = 0.0f; hlast[j] = 0.0f; }

  const __bf16* xrow = xbf + (size_t)brow * SEQ * INF;
  __syncthreads();

  for (int s = 0; s < SEQ; ++s) {
    const __bf16* hprev = hbf + (size_t)(s & 1) * BSZ * HS;
    __bf16*       hnew  = hbf + (size_t)((s + 1) & 1) * BSZ * HS;

    // ---- per-step fill: the 2 waves sharing mt stage its 24 A k-tiles ----
    // Each lane's slot = two contiguous 16B chunks -> 2 async b128 per tile.
    {
      const __bf16* xr = xrow + (size_t)s * INF;
      const __bf16* hr = hprev + (size_t)brow * HS;
#pragma unroll 4
      for (int t3 = 0; t3 < 12; ++t3) {
        int t = (int)hl * 12 + t3;
        const __bf16* rowp = (t < 8) ? (xr + t * 32 + khalf)
                                     : (hr + (t - 8) * 32 + khalf);
        if (t < 8) __builtin_prefetch(rowp + INF, 0, 0);  // next step's x
        __bf16* slot = &sA[((size_t)mt * 24 + t) * 512 + (size_t)lane * 16];
        copy16_g2l(rowp, slot);
        copy16_g2l(rowp + 8, slot + 8);
      }
    }
    wait_fill();
    __syncthreads();

    v8f acc[4];
#pragma unroll
    for (int g = 0; g < 4; ++g) {
#pragma unroll
      for (int j = 0; j < 8; ++j) acc[g][j] = bv[g];
    }

    // ---- x @ W_ih  (K = 256 -> 8 WMMA k-steps) ----
#pragma unroll 4
    for (int kk = 0; kk < INF / 32; ++kk) {
      v16bf a = *(const v16bf*)(&sA[((size_t)mt * 24 + kk) * 512 + (size_t)lane * 16]);
#pragma unroll
      for (int g = 0; g < 4; ++g) {
        const v16bf b = *(const v16bf*)(
            &sBih[(((size_t)hl * 4 + g) * 8 + kk) * 512 + (size_t)lane * 16]);
        acc[g] = __builtin_amdgcn_wmma_f32_16x16x32_bf16(
            false, a, false, b, (short)0, acc[g], false, false);
      }
    }

    // ---- h @ W_hh  (K = 512 -> 16 WMMA k-steps) ----
#pragma unroll 4
    for (int kk = 0; kk < HS / 32; ++kk) {
      v16bf a = *(const v16bf*)(&sA[((size_t)mt * 24 + 8 + kk) * 512 + (size_t)lane * 16]);
#pragma unroll
      for (int g = 0; g < 4; ++g) {
        const v16bf b = *(const v16bf*)(
            &sBhh[(((size_t)hl * 4 + g) * 16 + kk) * 512 + (size_t)lane * 16]);
        acc[g] = __builtin_amdgcn_wmma_f32_16x16x32_bf16(
            false, a, false, b, (short)0, acc[g], false, false);
      }
    }

    // ---- gate nonlinearities + state update (c stays in registers) ----
#pragma unroll
    for (int j = 0; j < 8; ++j) {
      float it = 1.0f / (1.0f + __expf(-acc[0][j]));
      float ft = 1.0f / (1.0f + __expf(-acc[1][j]));
      float gt = tanhf(acc[2][j]);
      float ot = 1.0f / (1.0f + __expf(-acc[3][j]));
      float cn = ft * c[j] + it * gt;
      c[j] = cn;
      float hn = ot * tanhf(cn);
      hlast[j] = hn;
      unsigned row = crow0 + (unsigned)j;                   // batch index
      hnew[(size_t)row * HS + col] = (__bf16)hn;            // next-step input
      out[((size_t)row * SEQ + (size_t)s) * HS + col] = hn; // hidden_seq f32
    }

    grid_barrier(bar, bar + 1);
  }

  // ---- final h_t, c_t ----
  float* h_t = out + (size_t)BSZ * SEQ * HS;
  float* c_t = h_t + (size_t)BSZ * HS;
#pragma unroll
  for (int j = 0; j < 8; ++j) {
    unsigned row = crow0 + (unsigned)j;
    h_t[(size_t)row * HS + col] = hlast[j];
    c_t[(size_t)row * HS + col] = c[j];
  }
}

// ---------------------------------------------------------------------------
extern "C" void kernel_launch(void* const* d_in, const int* in_sizes, int n_in,
                              void* d_out, int out_size, void* d_ws, size_t ws_size,
                              hipStream_t stream) {
  const float* x        = (const float*)d_in[0];
  const float* wih_mu   = (const float*)d_in[1];
  const float* wih_rho  = (const float*)d_in[2];
  const float* whh_mu   = (const float*)d_in[3];
  const float* whh_rho  = (const float*)d_in[4];
  const float* b_mu     = (const float*)d_in[5];
  const float* b_rho    = (const float*)d_in[6];
  const float* eps_ih   = (const float*)d_in[7];
  const float* eps_hh   = (const float*)d_in[8];
  const float* eps_b    = (const float*)d_in[9];

  unsigned char* ws = (unsigned char*)d_ws;
  unsigned* bar   = (unsigned*)(ws + WS_BAR);
  __bf16*   hbf   = (__bf16*)(ws + WS_HBF);
  float*    biasv = (float*)(ws + WS_BIAS);
  __bf16*   xbf   = (__bf16*)(ws + WS_XBF);
  __bf16*   pbih  = (__bf16*)(ws + WS_PIH);
  __bf16*   pbhh  = (__bf16*)(ws + WS_PHH);

  // zero barrier state + h[0] bf16 double buffer (graph-capture safe)
  (void)hipMemsetAsync(ws, 0, WS_BIAS, stream);

  sample_bias_kernel<<<(G4 + 255) / 256, 256, 0, stream>>>(
      b_mu, b_rho, eps_b, biasv, G4);
  cvt_x_kernel<<<(BSZ * SEQ * INF / 4 + 255) / 256, 256, 0, stream>>>(
      x, xbf, BSZ * SEQ * INF / 4);
  pack_weights_kernel<<<(INF * G4) / 256, 256, 0, stream>>>(
      wih_mu, wih_rho, eps_ih, pbih, INF, G4);
  pack_weights_kernel<<<(HS * G4) / 256, 256, 0, stream>>>(
      whh_mu, whh_rho, eps_hh, pbhh, HS, G4);

  lstm_scan_kernel<<<NWG, 256, 0, stream>>>(
      xbf, pbih, pbhh, biasv, hbf, bar, (float*)d_out);
}
